// MultiHeadAttention_22299470200954
// MI455X (gfx1250) — matrix operand order repaired and checked by reading, not compiled, not run on hardware
//
#include <hip/hip_runtime.h>

// ---------------- problem constants ----------------
#define B_  2
#define S_  2048
#define D_  1024
#define H_  16
#define DK_ 64

// ---------------- WMMA types ----------------
typedef __bf16          v16bf __attribute__((ext_vector_type(16)));
typedef float           v8f   __attribute__((ext_vector_type(8)));
typedef unsigned short  u16x8 __attribute__((ext_vector_type(8)));

typedef __attribute__((address_space(3))) unsigned short lds_ushort;

union FragBF {
    v16bf bf;
    u16x8 h[2];   // h[0] = VGPRs 0..3 (elements 0..7), h[1] = VGPRs 4..7 (elements 8..15)
};

__device__ __forceinline__ unsigned short f2bf(float f) {
    unsigned int u = __float_as_uint(f);
    u += 0x7FFFu + ((u >> 16) & 1u);      // round-to-nearest-even
    return (unsigned short)(u >> 16);
}

__device__ __forceinline__ v8f wmma_bf16(const FragBF& a, const FragBF& b, v8f c) {
    // D = A(16x32 bf16) * B(32x16 bf16) + C(16x16 f32)
    return __builtin_amdgcn_wmma_f32_16x16x32_bf16(
        /*neg_a=*/false, a.bf, /*neg_b=*/false, b.bf,
        /*c_mod=*/(short)0, c, /*reuse_a=*/false, /*reuse_b=*/false);
}

// Async 16-byte global -> LDS copy (per lane). Tracked by ASYNCcnt.
__device__ __forceinline__ void async_g2l_b128(unsigned lds_byte_off,
                                               const unsigned short* gptr) {
    asm volatile("global_load_async_to_lds_b128 %0, %1, off"
                 :: "v"(lds_byte_off), "v"(gptr)
                 : "memory");
}
__device__ __forceinline__ void wait_asynccnt0() {
    asm volatile("s_wait_asynccnt 0x0" ::: "memory");
}
__device__ __forceinline__ unsigned lds_offset(const unsigned short* p) {
    return (unsigned)(uintptr_t)(lds_ushort*)p;   // generic -> LDS addrspace(3) offset
}

// ---------------- fp32 -> bf16 convert ----------------
__global__ __launch_bounds__(256) void cvt_f32_bf16(const float* __restrict__ src,
                                                    unsigned short* __restrict__ dst,
                                                    int n4) {
    int i = blockIdx.x * blockDim.x + threadIdx.x;
    if (i < n4) {
        float4 v = ((const float4*)src)[i];
        ushort4 o;
        o.x = f2bf(v.x); o.y = f2bf(v.y); o.z = f2bf(v.z); o.w = f2bf(v.w);
        ((ushort4*)dst)[i] = o;
    }
}

// ---------------- GEMM mainloop: per-wave 32x64 tile, C = A * W^T ----------------
// A: MxK bf16 row-major.  W: NxK bf16 row-major (so B[k][n] = W[n][k]).
// A-frag (16x32, MxK): lane l (m=l%16, h=l/16) holds K = {8h..8h+7, 16+8h..16+8h+7}
// B-frag (32x16, KxN): lane l (n=l%16, h=l/16) holds K = {8h..8h+7, 16+8h..16+8h+7}
__device__ __forceinline__ void gemm_core_32x64(const unsigned short* __restrict__ A,
                                                const unsigned short* __restrict__ W,
                                                int K, int m0, int n0,
                                                int lm, int lh, v8f C[2][4]) {
    for (int kb = 0; kb < K; kb += 32) {
        FragBF fa[2], fb[4];
#pragma unroll
        for (int mi = 0; mi < 2; ++mi) {
            const unsigned short* p = A + (size_t)(m0 + mi * 16 + lm) * K + kb + 8 * lh;
            fa[mi].h[0] = *(const u16x8*)p;
            fa[mi].h[1] = *(const u16x8*)(p + 16);
        }
#pragma unroll
        for (int ni = 0; ni < 4; ++ni) {
            const unsigned short* p = W + (size_t)(n0 + ni * 16 + lm) * K + kb + 8 * lh;
            fb[ni].h[0] = *(const u16x8*)p;
            fb[ni].h[1] = *(const u16x8*)(p + 16);
        }
#pragma unroll
        for (int mi = 0; mi < 2; ++mi)
#pragma unroll
            for (int ni = 0; ni < 4; ++ni)
                C[mi][ni] = wmma_bf16(fa[mi], fb[ni], C[mi][ni]);
    }
}

// ---------------- QKV projection: y = x @ W.T + b, scattered to [B,H,S,DK] bf16 ----------------
__global__ __launch_bounds__(256) void qkv_gemm(
    const unsigned short* __restrict__ Xb,
    const unsigned short* __restrict__ Wq, const unsigned short* __restrict__ Wk,
    const unsigned short* __restrict__ Wv,
    const float* __restrict__ bq, const float* __restrict__ bk, const float* __restrict__ bv,
    unsigned short* __restrict__ Qb, unsigned short* __restrict__ Kb,
    unsigned short* __restrict__ Vb) {
    const int tid  = threadIdx.x;
    const int wave = tid >> 5, lane = tid & 31;
    const int lm = lane & 15, lh = lane >> 4;
    const int m0 = blockIdx.y * 128 + (wave & 3) * 32;
    const int n0 = blockIdx.x * 128 + (wave >> 2) * 64;
    const int z  = blockIdx.z;

    const unsigned short* W    = (z == 0) ? Wq : (z == 1) ? Wk : Wv;
    const float*          bias = (z == 0) ? bq : (z == 1) ? bk : bv;
    unsigned short*       dst  = (z == 0) ? Qb : (z == 1) ? Kb : Vb;

    v8f C[2][4];
#pragma unroll
    for (int mi = 0; mi < 2; ++mi)
#pragma unroll
        for (int ni = 0; ni < 4; ++ni) C[mi][ni] = {};

    gemm_core_32x64(Xb, W, D_, m0, n0, lm, lh, C);

    // C layout: VGPR r -> row (r + 8*(lane/16)), lane%16 -> col
#pragma unroll
    for (int mi = 0; mi < 2; ++mi)
#pragma unroll
        for (int ni = 0; ni < 4; ++ni)
#pragma unroll
            for (int r = 0; r < 8; ++r) {
                const int gm = m0 + mi * 16 + r + 8 * lh;   // row in [B*S]
                const int gn = n0 + ni * 16 + lm;           // col in [D]
                float val = C[mi][ni][r] + bias[gn];
                const int bb = gm >> 11, ss = gm & (S_ - 1);
                const int hh = gn >> 6,  dd = gn & (DK_ - 1);
                dst[(((size_t)bb * H_ + hh) * S_ + ss) * DK_ + dd] = f2bf(val);
            }
}

// ---------------- Flash attention: per (b,h), 128 query rows / block ----------------
// Double-buffered K/V staging: K via async global->LDS DMA, V via regs + transposed
// ds stores; block kb+1 streams in while block kb's WMMAs run.
__global__ __launch_bounds__(256) void attn_kernel(
    const unsigned short* __restrict__ Qb,
    const unsigned short* __restrict__ Kb,
    const unsigned short* __restrict__ Vb,
    unsigned short* __restrict__ Ob) {
    __shared__ alignas(16) unsigned short sK[2][32 * 64];     // [buf][key][d]
    __shared__ alignas(16) unsigned short sVt[2][64 * 32];    // [buf][d][key] (transposed)
    __shared__ alignas(16) unsigned short sP[8][16 * 32];     // per-wave P scratch

    const int tid  = threadIdx.x;
    const int wave = tid >> 5, lane = tid & 31;
    const int lm = lane & 15, lh = lane >> 4;
    const int q0 = blockIdx.x * 128;
    const int h  = blockIdx.y, b = blockIdx.z;
    const int bh = b * H_ + h;
    const int qrow = q0 + wave * 16;

    const unsigned short* Qh = Qb + (size_t)bh * S_ * DK_;
    const unsigned short* Kh = Kb + (size_t)bh * S_ * DK_;
    const unsigned short* Vh = Vb + (size_t)bh * S_ * DK_;

    // Q fragments for d = [0,32) and [32,64)
    FragBF aq[2];
    {
        const unsigned short* qp = Qh + (size_t)(qrow + lm) * DK_;
#pragma unroll
        for (int c = 0; c < 2; ++c) {
            aq[c].h[0] = *(const u16x8*)(qp + c * 32 + 8 * lh);
            aq[c].h[1] = *(const u16x8*)(qp + c * 32 + 16 + 8 * lh);
        }
    }

    v8f   Oacc[4];
    float mrow[8], lrow[8];
#pragma unroll
    for (int t = 0; t < 4; ++t) Oacc[t] = {};
#pragma unroll
    for (int r = 0; r < 8; ++r) { mrow[r] = -1e30f; lrow[r] = 0.0f; }

    const int numKB = q0 / 32 + 4;         // 32-key blocks covering causal range of block
    const int ckey  = tid >> 3;            // cooperative load: key row 0..31
    const int cd    = (tid & 7) * 8;       // d chunk 0..56

    // ---- pipeline prologue: issue block 0 ----
    u16x8 vreg = *(const u16x8*)(Vh + (size_t)ckey * DK_ + cd);          // V(0) -> regs
    async_g2l_b128(lds_offset(&sK[0][ckey * 64 + cd]),
                   Kh + (size_t)ckey * DK_ + cd);                        // K(0) -> LDS (async)

    int cur = 0;
    for (int kb = 0; kb < numKB; ++kb) {
        const int kbase = kb * 32;
        // publish V(kb): transposed store of the regs loaded last iteration
#pragma unroll
        for (int i = 0; i < 8; ++i) sVt[cur][(cd + i) * 32 + ckey] = vreg[i];
        wait_asynccnt0();            // my K(kb) async chunk landed in LDS
        __syncthreads();             // all waves' K/V chunks visible

        // ---- stream block kb+1 into the other buffer while we compute ----
        if (kb + 1 < numKB) {
            const int nb = kbase + 32;
            vreg = *(const u16x8*)(Vh + (size_t)(nb + ckey) * DK_ + cd);
            async_g2l_b128(lds_offset(&sK[cur ^ 1][ckey * 64 + cd]),
                           Kh + (size_t)(nb + ckey) * DK_ + cd);
        }

        if (kbase <= qrow + 15) {
            // S = Q @ K^T for two 16-key subtiles
            v8f s[2];
#pragma unroll
            for (int ts = 0; ts < 2; ++ts) {
                v8f acc = {};
#pragma unroll
                for (int c = 0; c < 2; ++c) {
                    FragBF fk;
                    const unsigned short* p = &sK[cur][(ts * 16 + lm) * 64 + c * 32 + lh * 8];
                    fk.h[0] = *(const u16x8*)p;
                    fk.h[1] = *(const u16x8*)(p + 16);
                    acc = wmma_bf16(aq[c], fk, acc);
                }
                s[ts] = acc;
            }
            // scale + causal mask
#pragma unroll
            for (int ts = 0; ts < 2; ++ts) {
                const int gk = kbase + ts * 16 + lm;
#pragma unroll
                for (int r = 0; r < 8; ++r) {
                    const int gq = qrow + r + 8 * lh;
                    float v = s[ts][r] * 0.125f;      // 1/sqrt(64)
                    s[ts][r] = (gk <= gq) ? v : -1e30f;
                }
            }
            // online softmax (row r lives in VGPR r across a 16-lane half)
            float alpha[8];
#pragma unroll
            for (int r = 0; r < 8; ++r) {
                float bm = fmaxf(s[0][r], s[1][r]);
                bm = fmaxf(bm, __shfl_xor(bm, 1, 32));
                bm = fmaxf(bm, __shfl_xor(bm, 2, 32));
                bm = fmaxf(bm, __shfl_xor(bm, 4, 32));
                bm = fmaxf(bm, __shfl_xor(bm, 8, 32));
                float nm = fmaxf(mrow[r], bm);
                alpha[r] = __expf(mrow[r] - nm);
                mrow[r]  = nm;
                float p0 = __expf(s[0][r] - nm);
                float p1 = __expf(s[1][r] - nm);
                s[0][r] = p0; s[1][r] = p1;
                float rs = p0 + p1;
                rs += __shfl_xor(rs, 1, 32);
                rs += __shfl_xor(rs, 2, 32);
                rs += __shfl_xor(rs, 4, 32);
                rs += __shfl_xor(rs, 8, 32);
                lrow[r] = lrow[r] * alpha[r] + rs;
            }
#pragma unroll
            for (int t = 0; t < 4; ++t)
#pragma unroll
                for (int r = 0; r < 8; ++r) Oacc[t][r] *= alpha[r];

            // re-swizzle P: C-layout -> A-fragment layout through wave-private LDS
            unsigned short* ps = sP[wave];
#pragma unroll
            for (int r = 0; r < 8; ++r) {
                const int m_ = r + 8 * lh;
                ps[m_ * 32 + lm]      = f2bf(s[0][r]);
                ps[m_ * 32 + 16 + lm] = f2bf(s[1][r]);
            }
            FragBF ap;
            {
                const unsigned short* pp = ps + lm * 32;
                ap.h[0] = *(const u16x8*)(pp + 8 * lh);
                ap.h[1] = *(const u16x8*)(pp + 16 + 8 * lh);
            }
            // O += P @ V  (B-frags contiguous from transposed V)
#pragma unroll
            for (int t = 0; t < 4; ++t) {
                FragBF fv;
                const unsigned short* vp = &sVt[cur][(t * 16 + lm) * 32 + lh * 8];
                fv.h[0] = *(const u16x8*)vp;
                fv.h[1] = *(const u16x8*)(vp + 16);
                Oacc[t] = wmma_bf16(ap, fv, Oacc[t]);
            }
        }
        cur ^= 1;
    }

    // finalize: divide by row sums, write bf16 [B,H,S,DK] flat (== reference reshape)
    unsigned short* op = Ob + (size_t)bh * S_ * DK_;
#pragma unroll
    for (int r = 0; r < 8; ++r) {
        const int m_ = r + 8 * lh;
        const float rl = 1.0f / lrow[r];
#pragma unroll
        for (int t = 0; t < 4; ++t)
            op[(size_t)(qrow + m_) * DK_ + t * 16 + lm] = f2bf(Oacc[t][r] * rl);
    }
}

// ---------------- output projection: out = O @ wo.T + bo (fp32 out) ----------------
__global__ __launch_bounds__(256) void out_gemm(
    const unsigned short* __restrict__ Ab,
    const unsigned short* __restrict__ Wo,
    const float* __restrict__ bo,
    float* __restrict__ out) {
    const int tid  = threadIdx.x;
    const int wave = tid >> 5, lane = tid & 31;
    const int lm = lane & 15, lh = lane >> 4;
    const int m0 = blockIdx.y * 128 + (wave & 3) * 32;
    const int n0 = blockIdx.x * 128 + (wave >> 2) * 64;

    v8f C[2][4];
#pragma unroll
    for (int mi = 0; mi < 2; ++mi)
#pragma unroll
        for (int ni = 0; ni < 4; ++ni) C[mi][ni] = {};

    gemm_core_32x64(Ab, Wo, D_, m0, n0, lm, lh, C);

#pragma unroll
    for (int mi = 0; mi < 2; ++mi)
#pragma unroll
        for (int ni = 0; ni < 4; ++ni)
#pragma unroll
            for (int r = 0; r < 8; ++r) {
                const int gm = m0 + mi * 16 + r + 8 * lh;
                const int gn = n0 + ni * 16 + lm;
                out[(size_t)gm * D_ + gn] = C[mi][ni][r] + bo[gn];
            }
}

// ---------------- launch ----------------
extern "C" void kernel_launch(void* const* d_in, const int* in_sizes, int n_in,
                              void* d_out, int out_size, void* d_ws, size_t ws_size,
                              hipStream_t stream) {
    (void)in_sizes; (void)n_in; (void)out_size; (void)ws_size;
    const float* x  = (const float*)d_in[0];
    const float* wq = (const float*)d_in[1];
    const float* bq = (const float*)d_in[2];
    const float* wk = (const float*)d_in[3];
    const float* bk = (const float*)d_in[4];
    const float* wv = (const float*)d_in[5];
    const float* bv = (const float*)d_in[6];
    const float* wo = (const float*)d_in[7];
    const float* bo = (const float*)d_in[8];
    float* out = (float*)d_out;

    char* ws = (char*)d_ws;
    unsigned short* xb  = (unsigned short*)(ws);                        // 8 MB
    unsigned short* wqb = (unsigned short*)(ws + ((size_t)8  << 20));   // 2 MB
    unsigned short* wkb = (unsigned short*)(ws + ((size_t)10 << 20));
    unsigned short* wvb = (unsigned short*)(ws + ((size_t)12 << 20));
    unsigned short* wob = (unsigned short*)(ws + ((size_t)14 << 20));
    unsigned short* Qb  = (unsigned short*)(ws + ((size_t)16 << 20));   // 8 MB each
    unsigned short* Kb  = (unsigned short*)(ws + ((size_t)24 << 20));
    unsigned short* Vb  = (unsigned short*)(ws + ((size_t)32 << 20));
    unsigned short* Ob  = (unsigned short*)(ws + ((size_t)40 << 20));

    // 1) fp32 -> bf16
    cvt_f32_bf16<<<4096, 256, 0, stream>>>(x,  xb,  (B_ * S_ * D_) / 4);
    cvt_f32_bf16<<<1024, 256, 0, stream>>>(wq, wqb, (D_ * D_) / 4);
    cvt_f32_bf16<<<1024, 256, 0, stream>>>(wk, wkb, (D_ * D_) / 4);
    cvt_f32_bf16<<<1024, 256, 0, stream>>>(wv, wvb, (D_ * D_) / 4);
    cvt_f32_bf16<<<1024, 256, 0, stream>>>(wo, wob, (D_ * D_) / 4);

    // 2) fused QKV projections (z selects Q/K/V)
    qkv_gemm<<<dim3(D_ / 128, (B_ * S_) / 128, 3), 256, 0, stream>>>(
        xb, wqb, wkb, wvb, bq, bk, bv, Qb, Kb, Vb);

    // 3) causal flash attention (async-LDS double-buffered K/V)
    attn_kernel<<<dim3(S_ / 128, H_, B_), 256, 0, stream>>>(Qb, Kb, Vb, Ob);

    // 4) output projection
    out_gemm<<<dim3(D_ / 128, (B_ * S_) / 128), 256, 0, stream>>>(Ob, wob, bo, out);
}
